// GeometricCPE_12086037971113
// MI455X (gfx1250) — matrix operand order repaired and checked by reading, not compiled, not run on hardware
//
#include <hip/hip_runtime.h>

typedef __attribute__((ext_vector_type(16))) __bf16 v16bf;
typedef __attribute__((ext_vector_type(8)))  float  v8f;

constexpr int BQ = 256;   // batches
constexpr int NQ = 128;   // particles
constexpr int CQ = 256;   // channels
constexpr int HM = 40;    // grid H
constexpr int WM = 40;    // grid W
constexpr int CHUNK = 8;  // channels per LDS grid pass (40*40*8*4 = 50KB LDS)

// ---------------------------------------------------------------------------
// Phase 0: per-batch min of eta/phi, quantize to int grid indices.
// ---------------------------------------------------------------------------
__global__ __launch_bounds__(128) void quant_kernel(
    const float* __restrict__ eta, const float* __restrict__ phi,
    int* __restrict__ ge, int* __restrict__ gp)
{
    int b = blockIdx.x, n = threadIdx.x;
    float e = eta[b * NQ + n];
    float p = phi[b * NQ + n];
    __shared__ float se[NQ], sp[NQ];
    se[n] = e; sp[n] = p;
    __syncthreads();
    for (int s = NQ / 2; s > 0; s >>= 1) {
        if (n < s) {
            se[n] = fminf(se[n], se[n + s]);
            sp[n] = fminf(sp[n], sp[n + s]);
        }
        __syncthreads();
    }
    float emin = se[0], pmin = sp[0];
    ge[b * NQ + n] = (int)((e - emin) / 0.05f);
    gp[b * NQ + n] = (int)((p - pmin) / 0.05f);
}

// ---------------------------------------------------------------------------
// Phase 0b: pre-swizzle dense_w (f32, KxN row-major) into bf16 WMMA B-fragments.
// Fragment layout (contiguous k order chosen here, same as A; 16-bit B 32x16): element i of lane L:
//   N = 16*nt + (L%16),  K = 32*kk + 16*(L>=16) + i
// Stored K-slice-major so each kk slice is one contiguous 16KB block:
//   bp[(((kk*16 + nt)*32 + lane)*16 + i]
// ---------------------------------------------------------------------------
__global__ __launch_bounds__(256) void pack_b_kernel(
    const float* __restrict__ w, __bf16* __restrict__ bp)
{
    int gid  = blockIdx.x * 256 + threadIdx.x;   // 8*16*32*16 = 65536 elements
    int idx  = gid & 15;
    int lane = (gid >> 4) & 31;
    int nt   = (gid >> 9) & 15;
    int kk   = gid >> 13;
    int K = kk * 32 + ((lane >> 4) << 4) + idx;
    int N = nt * 16 + (lane & 15);
    bp[gid] = (__bf16)w[K * CQ + N];
}

// ---------------------------------------------------------------------------
// Phase 1: LDS-resident grid scatter + depthwise 3x3 conv + gather.
// The block's 4KB x-chunk is fetched with global_load_async_to_lds while the
// 50KB LDS grid is being zeroed. Writes y into d_out (phase 2 overwrites).
// ---------------------------------------------------------------------------
__global__ __launch_bounds__(256) void convgather_kernel(
    const float* __restrict__ x, const int* __restrict__ ge, const int* __restrict__ gp,
    const float* __restrict__ ck, const float* __restrict__ cb, float* __restrict__ y)
{
    const int b   = blockIdx.y;
    const int c0  = blockIdx.x * CHUNK;
    const int tid = threadIdx.x;

    __shared__ float grid[HM * WM * CHUNK];              // 50 KB
    __shared__ __align__(16) float sx[NQ * CHUNK];       // 4 KB staged x chunk
    __shared__ int   sge[NQ], sgp[NQ];
    __shared__ float sk[9 * CHUNK], sb[CHUNK];

    // async-copy this block's x chunk into LDS (overlaps with grid zeroing)
    {
        int n = tid >> 1, q = tid & 1;   // 16B per thread = 4 channels
        uint64_t g = (uint64_t)(uintptr_t)(x + ((size_t)b * NQ + n) * CQ + c0 + q * 4);
        unsigned  l = (unsigned)(uintptr_t)sx + (unsigned)tid * 16u;
        asm volatile("global_load_async_to_lds_b128 %0, %1, off"
                     :: "v"(l), "v"(g) : "memory");
    }

    for (int i = tid; i < HM * WM * CHUNK; i += 256) grid[i] = 0.0f;
    if (tid < NQ) { sge[tid] = ge[b * NQ + tid]; sgp[tid] = gp[b * NQ + tid]; }
    if (tid < 9 * CHUNK) {                        // conv_kernel [3][3][1][C]
        int t = tid / CHUNK, c = tid % CHUNK;
        sk[tid] = ck[t * CQ + c0 + c];
    }
    if (tid < CHUNK) sb[tid] = cb[c0 + tid];

    asm volatile("s_wait_asynccnt 0x0" ::: "memory");
    __syncthreads();

    // scatter-add particles into LDS grid (ds_add_f32)
#pragma unroll
    for (int it = 0; it < (NQ * CHUNK) / 256; ++it) {
        int idx = it * 256 + tid;
        int n = idx >> 3, c = idx & (CHUNK - 1);
        atomicAdd(&grid[(sge[n] * WM + sgp[n]) * CHUNK + c], sx[n * CHUNK + c]);
    }
    __syncthreads();

    // gather 3x3 neighborhood with conv taps (cross-correlation, SAME zero pad)
#pragma unroll
    for (int it = 0; it < (NQ * CHUNK) / 256; ++it) {
        int idx = it * 256 + tid;
        int n = idx >> 3, c = idx & (CHUNK - 1);
        int gh = sge[n], gw = sgp[n];
        float acc = sb[c];
#pragma unroll
        for (int dh = 0; dh < 3; ++dh) {
            int h = gh + dh - 1;
            if ((unsigned)h < (unsigned)HM) {
#pragma unroll
                for (int dw = 0; dw < 3; ++dw) {
                    int w = gw + dw - 1;
                    if ((unsigned)w < (unsigned)WM)
                        acc += sk[(dh * 3 + dw) * CHUNK + c] * grid[(h * WM + w) * CHUNK + c];
                }
            }
        }
        y[((size_t)b * NQ + n) * CQ + c0 + c] = acc;
    }
}

// ---------------------------------------------------------------------------
// Phase 2: per-batch GEMM (128x256 @ 256x256) via v_wmma_f32_16x16x32_bf16.
// B K-slices (16KB each) are double-buffered through LDS with
// global_load_async_to_lds_b128; waves read fragments via ds_load_b128.
// Fused dense bias + LayerNorm (shfl_xor over 16-lane column group) + residual.
// Reads y from d_out, overwrites d_out (each wave owns its 16 rows).
// ---------------------------------------------------------------------------
__device__ __forceinline__ void async_copy_bslice(
    const __bf16* bp, int kk, unsigned lds_base, int tid)
{
#pragma unroll
    for (int i = 0; i < 4; ++i) {   // 256 thr * 4 * 16B = 16KB slice
        unsigned  l = lds_base + (unsigned)((i * 256 + tid) * 16);
        uint64_t  g = (uint64_t)(uintptr_t)((const char*)bp
                     + (size_t)kk * 16384 + (size_t)(i * 256 + tid) * 16);
        asm volatile("global_load_async_to_lds_b128 %0, %1, off"
                     :: "v"(l), "v"(g) : "memory");
    }
}

__global__ __launch_bounds__(256) void gemm_ln_kernel(
    const float* __restrict__ x, const float* __restrict__ y,
    const __bf16* __restrict__ bp, const float* __restrict__ db,
    const float* __restrict__ gamma, const float* __restrict__ beta,
    float* __restrict__ out)
{
    const int b     = blockIdx.x;
    const int tid   = threadIdx.x;
    const int wave  = tid >> 5;
    const int lane  = tid & 31;
    const int laneN = lane & 15;
    const int hi    = lane >> 4;

    __shared__ __align__(32) __bf16 bbuf[2][16 * 32 * 16];   // 2 x 16KB K-slices

    const float* yb  = y + (size_t)b * NQ * CQ;
    const size_t base = (size_t)b * NQ * CQ;

    // prefetch residual-x block (128KB) into cache for the epilogue
    const char* px = (const char*)(x + base);
#pragma unroll
    for (int i = 0; i < 4; ++i)
        __builtin_prefetch(px + (size_t)(i * 256 + tid) * 128, 0, 1);

    const unsigned lbase = (unsigned)(uintptr_t)(&bbuf[0][0]);
    async_copy_bslice(bp, 0, lbase, tid);

    v8f acc[16] = {};                 // 16 N-tiles x 8 f32 accum VGPRs
    const int arow = wave * 16 + laneN;

#pragma unroll
    for (int kk = 0; kk < 8; ++kk) {
        // A fragment (16x32 bf16): lane holds row `arow`,
        // K = 32*kk + 16*hi + {0..15}   (issue loads early)
        const float* ap = yb + (size_t)arow * CQ + kk * 32 + hi * 16;
        float4 a0 = *(const float4*)(ap + 0);
        float4 a1 = *(const float4*)(ap + 4);
        float4 a2 = *(const float4*)(ap + 8);
        float4 a3 = *(const float4*)(ap + 12);

        // kick off next K-slice copy, then wait for current slice only
        if (kk < 7) {
            async_copy_bslice(bp, kk + 1, lbase + (unsigned)(((kk + 1) & 1) * 16384), tid);
            asm volatile("s_wait_asynccnt 0x4" ::: "memory");
        } else {
            asm volatile("s_wait_asynccnt 0x0" ::: "memory");
        }
        __syncthreads();

        v16bf af;
        af[0]  = (__bf16)a0.x; af[1]  = (__bf16)a0.y; af[2]  = (__bf16)a0.z; af[3]  = (__bf16)a0.w;
        af[4]  = (__bf16)a1.x; af[5]  = (__bf16)a1.y; af[6]  = (__bf16)a1.z; af[7]  = (__bf16)a1.w;
        af[8]  = (__bf16)a2.x; af[9]  = (__bf16)a2.y; af[10] = (__bf16)a2.z; af[11] = (__bf16)a2.w;
        af[12] = (__bf16)a3.x; af[13] = (__bf16)a3.y; af[14] = (__bf16)a3.z; af[15] = (__bf16)a3.w;

        const v16bf* bsl = (const v16bf*)(&bbuf[kk & 1][0]);
#pragma unroll
        for (int nt = 0; nt < 16; ++nt) {
            v16bf bf = bsl[nt * 32 + lane];          // ds_load_b128 x2
            acc[nt] = __builtin_amdgcn_wmma_f32_16x16x32_bf16(
                false, af, false, bf, (short)0, acc[nt], false, false);
        }
        __syncthreads();   // slice consumed; safe to overwrite next round
    }

    // dense bias (before LN stats, as in reference)
#pragma unroll
    for (int nt = 0; nt < 16; ++nt) {
        float bias = db[nt * 16 + laneN];
#pragma unroll
        for (int v = 0; v < 8; ++v) acc[nt][v] += bias;
    }

    // LayerNorm stats per row. C/D layout: element v of lane -> row 16w+v+8*hi,
    // col 16*nt+laneN. Reduce over the 16-lane column group (xor 1,2,4,8).
    float mean[8], rs[8];
#pragma unroll
    for (int v = 0; v < 8; ++v) {
        float s = 0.0f, q = 0.0f;
#pragma unroll
        for (int nt = 0; nt < 16; ++nt) { float t = acc[nt][v]; s += t; q += t * t; }
#pragma unroll
        for (int m = 8; m >= 1; m >>= 1) {
            s += __shfl_xor(s, m, 32);
            q += __shfl_xor(q, m, 32);
        }
        float mu  = s * (1.0f / CQ);
        float var = q * (1.0f / CQ) - mu * mu;
        mean[v] = mu;
        rs[v]   = rsqrtf(var + 1e-6f);
    }

    // normalize * gamma + beta, residual add, store
#pragma unroll
    for (int nt = 0; nt < 16; ++nt) {
        int col = nt * 16 + laneN;
        float g  = gamma[col];
        float be = beta[col];
#pragma unroll
        for (int v = 0; v < 8; ++v) {
            int row = wave * 16 + v + hi * 8;
            size_t o = base + (size_t)row * CQ + col;
            out[o] = x[o] + (acc[nt][v] - mean[v]) * rs[v] * g + be;
        }
    }
}

// ---------------------------------------------------------------------------
extern "C" void kernel_launch(void* const* d_in, const int* in_sizes, int n_in,
                              void* d_out, int out_size, void* d_ws, size_t ws_size,
                              hipStream_t stream)
{
    const float* x     = (const float*)d_in[0];
    const float* eta   = (const float*)d_in[1];
    const float* phi   = (const float*)d_in[2];
    const float* ck    = (const float*)d_in[3];
    const float* cb    = (const float*)d_in[4];
    const float* dw    = (const float*)d_in[5];
    const float* db    = (const float*)d_in[6];
    const float* gamma = (const float*)d_in[7];
    const float* beta  = (const float*)d_in[8];
    float* out = (float*)d_out;

    char* ws = (char*)d_ws;
    int*    ge = (int*)ws;                         // 256*128*4 = 128 KB
    int*    gp = (int*)(ws + 131072);              // 128 KB
    __bf16* bp = (__bf16*)(ws + 262144);           // 65536 bf16 = 128 KB

    quant_kernel<<<BQ, NQ, 0, stream>>>(eta, phi, ge, gp);
    pack_b_kernel<<<256, 256, 0, stream>>>(dw, bp);
    convgather_kernel<<<dim3(CQ / CHUNK, BQ), 256, 0, stream>>>(x, ge, gp, ck, cb, out);
    gemm_ln_kernel<<<BQ, 256, 0, stream>>>(x, out, bp, db, gamma, beta, out);
}
